// myLSTM_88983132438966
// MI455X (gfx1250) — matrix operand order repaired and checked by reading, not compiled, not run on hardware
//
#include <hip/hip_runtime.h>

// ---------------------------------------------------------------------------
// Problem dims (from reference): B=64, T=96, F=2048, H=2048, G=4H=8192, NC=2
// ---------------------------------------------------------------------------
#define BB   64
#define TT   96
#define FF   2048
#define HH   2048
#define GG   8192            // 4*H
#define KK   4096            // F + H (concatenated GEMM K)
#define KSPLIT 4
#define KCHUNK (KK / KSPLIT) // 1024: splits 0-1 entirely in x, 2-3 entirely in h

typedef __attribute__((ext_vector_type(16))) __bf16 v16bf;
typedef __attribute__((ext_vector_type(8)))  __bf16 v8bf;
typedef __attribute__((ext_vector_type(8)))  float  v8f;

__device__ __forceinline__ __bf16 f2bf(float f) {
    union { float f; unsigned u; } v; v.f = f;
    unsigned r = v.u + 0x7FFFu + ((v.u >> 16) & 1u);   // round-to-nearest-even
    unsigned short hs = (unsigned short)(r >> 16);
    union { unsigned short s; __bf16 b; } o; o.s = hs;
    return o.b;
}

__device__ __forceinline__ float sigm(float x) { return 1.0f / (1.0f + expf(-x)); }

__device__ __forceinline__ float u01_hash(unsigned x) {
    x ^= x >> 17; x *= 0xED5AD4BBu;
    x ^= x >> 11; x *= 0xAC4C1B51u;
    x ^= x >> 15; x *= 0x31848BABu;
    x ^= x >> 14;
    return ((float)(x >> 8) + 0.5f) * (1.0f / 16777216.0f);
}

// ---------------------------------------------------------------------------
// Conversion / init kernels
// ---------------------------------------------------------------------------

// Wcat[g][0:F] = W_ih[g][:], Wcat[g][F:F+H] = W_hh[g][:]   (bf16)
__global__ void cvt_wcat_kernel(const float* __restrict__ Wih,
                                const float* __restrict__ Whh,
                                __bf16* __restrict__ Wcat) {
    unsigned idx = blockIdx.x * 256u + threadIdx.x;      // [0, GG*KK)
    unsigned g = idx >> 12;                              // /4096
    unsigned k = idx & 4095u;
    float s = (k < FF) ? Wih[(size_t)g * FF + k] : Whh[(size_t)g * HH + (k - FF)];
    Wcat[idx] = f2bf(s);
}

// xbf[(t*B+b)*F + f] = bf16(feature[(b*T+t)*F + f])
__global__ void cvt_x_kernel(const float* __restrict__ feat,
                             __bf16* __restrict__ xbf) {
    unsigned idx = blockIdx.x * 256u + threadIdx.x;      // [0, T*B*F)
    unsigned f  = idx & (FF - 1);
    unsigned tb = idx >> 11;
    unsigned t = tb / BB, b = tb % BB;
    xbf[idx] = f2bf(feat[((size_t)b * TT + t) * FF + f]);
}

// bsum[g] = b_ih[g] + b_hh[g];  zero h, c, hbf
__global__ void init_state_kernel(const float* __restrict__ bih,
                                  const float* __restrict__ bhh,
                                  float* __restrict__ bsum,
                                  float* __restrict__ h, float* __restrict__ c,
                                  __bf16* __restrict__ hbf) {
    unsigned idx = blockIdx.x * 256u + threadIdx.x;      // grid covers 131072
    if (idx < GG) bsum[idx] = bih[idx] + bhh[idx];
    if (idx < BB * HH) {
        h[idx] = 0.0f; c[idx] = 0.0f;
        hbf[idx] = f2bf(0.0f);
    }
}

// ---------------------------------------------------------------------------
// Recurrent GEMM (split-K): part[ks](64 x 8192) = A_ks(64 x 1024) * Wcat_ks^T
// A = [x_t | h]; k-splits 0,1 read x_t, k-splits 2,3 read h (uniform base ptr).
// One wave -> 16(M) x 64(N) strip over a 1024-wide K range:
//   2048 waves/step; 32 K-chunks x 4 WMMA; ping-pong double buffer, fully
//   unrolled so all load addresses are base+immediate (no loop-carried adds,
//   no register rotation movs).
// A/B fragments (wave32): lane half h holds contiguous K = 16h + 0..15, same order in A and B.
// ---------------------------------------------------------------------------
__global__ void __launch_bounds__(256)
gemm_step_kernel(const __bf16* __restrict__ xbf_t,   // (64, 2048) bf16
                 const __bf16* __restrict__ hbf,     // (64, 2048) bf16
                 const __bf16* __restrict__ Wcat,    // (8192, 4096) bf16
                 float* __restrict__ part)           // (KSPLIT, 64, 8192)
{
    const unsigned lane = threadIdx.x & 31u;
    const unsigned w    = blockIdx.x * 8u + (threadIdx.x >> 5);   // [0, 2048)
    const unsigned ks    = w >> 9;                 // k-split [0,4)
    const unsigned strip = w & 511u;
    const unsigned tileM = (strip >> 7) << 4;      // 0,16,32,48
    const unsigned tileN = (strip & 127u) * 64u;   // 0..8128 step 64
    const unsigned half  = lane >> 4;              // 0/1
    const unsigned q     = lane & 15u;
    const unsigned kbase = ks * KCHUNK;

    // A row pointer for this lane (rows M=0..15 held by lanes q, both halves)
    const __bf16* arow = (ks < 2)
        ? (xbf_t + (size_t)(tileM + q) * FF + kbase + half * 16)
        : (hbf   + (size_t)(tileM + q) * HH + (kbase - FF) + half * 16);

    // B row pointers (one Wcat row per lane column, per 16-wide sub-tile)
    const __bf16* brow[4];
#pragma unroll
    for (int s = 0; s < 4; ++s)
        brow[s] = Wcat + (size_t)(tileN + s * 16 + q) * KK + kbase + half * 16;

    auto loadA = [&](unsigned k) -> v16bf {
        return *(const v16bf*)(arow + k);          // K = k + half*16 + 0..15
    };

    v8f acc[4];
#pragma unroll
    for (int s = 0; s < 4; ++s)
#pragma unroll
        for (int r = 0; r < 8; ++r) acc[s][r] = 0.0f;

    // ---- ping-pong pipeline, fully unrolled ----
    v16bf a0 = loadA(0);
    v16bf a1 = loadA(32);
    v16bf b0[4], b1[4];
#pragma unroll
    for (int s = 0; s < 4; ++s) {
        b0[s] = *(const v16bf*)(brow[s]);
        b1[s] = *(const v16bf*)(brow[s] + 32);
    }

#pragma unroll
    for (unsigned k = 0; k < KCHUNK - 64; k += 64) {
        // compute chunk k on buffer0, refill buffer0 with chunk k+64
#pragma unroll
        for (int s = 0; s < 4; ++s)
            acc[s] = __builtin_amdgcn_wmma_f32_16x16x32_bf16(
                false, a0, false, b0[s], (short)0, acc[s], false, false);
        a0 = loadA(k + 64);
#pragma unroll
        for (int s = 0; s < 4; ++s) b0[s] = *(const v16bf*)(brow[s] + k + 64);

        // compute chunk k+32 on buffer1, refill buffer1 with chunk k+96
#pragma unroll
        for (int s = 0; s < 4; ++s)
            acc[s] = __builtin_amdgcn_wmma_f32_16x16x32_bf16(
                false, a1, false, b1[s], (short)0, acc[s], false, false);
        a1 = loadA(k + 96);
#pragma unroll
        for (int s = 0; s < 4; ++s) b1[s] = *(const v16bf*)(brow[s] + k + 96);
    }
    // epilogue: last two chunks (KCHUNK-64, KCHUNK-32)
#pragma unroll
    for (int s = 0; s < 4; ++s)
        acc[s] = __builtin_amdgcn_wmma_f32_16x16x32_bf16(
            false, a0, false, b0[s], (short)0, acc[s], false, false);
#pragma unroll
    for (int s = 0; s < 4; ++s)
        acc[s] = __builtin_amdgcn_wmma_f32_16x16x32_bf16(
            false, a1, false, b1[s], (short)0, acc[s], false, false);

    // D layout: lane -> (M = half*8 + r, N = q)
    float* pp = part + (size_t)ks * BB * GG;
    const unsigned dm = tileM + half * 8;
#pragma unroll
    for (int s = 0; s < 4; ++s) {
        const unsigned n = tileN + s * 16 + q;
#pragma unroll
        for (int r = 0; r < 8; ++r)
            pp[(size_t)(dm + r) * GG + n] = acc[s][r];
    }
}

// ---------------------------------------------------------------------------
// LSTM cell elementwise (sums the split-K partials + bias):
//   c' = sig(f)*c + sig(i)*tanh(g); h' = sig(o)*tanh(c')
// ---------------------------------------------------------------------------
__global__ void cell_kernel(const float* __restrict__ part,   // (KSPLIT,64,8192)
                            const float* __restrict__ bsum,   // (8192)
                            float* __restrict__ h, float* __restrict__ c,
                            __bf16* __restrict__ hbf) {
    unsigned idx = blockIdx.x * 256u + threadIdx.x;      // [0, 64*2048)
    unsigned b = idx >> 11;
    unsigned j = idx & (HH - 1);
    const size_t PS = (size_t)BB * GG;
    const float* pb = part + (size_t)b * GG;

    auto gate = [&](unsigned g) -> float {
        return pb[g] + pb[PS + g] + pb[2 * PS + g] + pb[3 * PS + g] + bsum[g];
    };
    float gi = gate(j);
    float gf = gate(HH + j);
    float gg = gate(2 * HH + j);
    float go = gate(3 * HH + j);
    float cn = sigm(gf) * c[idx] + sigm(gi) * tanhf(gg);
    float hn = sigm(go) * tanhf(cn);
    c[idx] = cn;
    h[idx] = hn;
    hbf[idx] = f2bf(hn);
}

// ---------------------------------------------------------------------------
// Heads: per batch row, 7 dot products of length H, then bias/softmax, write out.
// out layout (floats): pred[T*B*2] | pred_c[T*B*2] | util[T*B] | watch[(T-1)*B*2]
// ---------------------------------------------------------------------------
__global__ void heads_kernel(const float* __restrict__ h, const float* __restrict__ c,
                             const float* __restrict__ Wp,  const float* __restrict__ bp,
                             const float* __restrict__ Wpc, const float* __restrict__ bpc,
                             const float* __restrict__ Wu,  const float* __restrict__ bu,
                             const float* __restrict__ Wuse,const float* __restrict__ buse,
                             float* __restrict__ out, int t) {
    const int b = blockIdx.x;
    const int tid = threadIdx.x;
    const float* hb = h + (size_t)b * HH;
    const float* cb = c + (size_t)b * HH;

    float s0 = 0.f, s1 = 0.f, s2 = 0.f, s3 = 0.f, s4 = 0.f, s5 = 0.f, s6 = 0.f;
    for (int j = tid; j < HH; j += 256) {
        float hv = hb[j], cv = cb[j];
        s0 += hv * Wp[j];        s1 += hv * Wp[HH + j];
        s2 += cv * Wpc[j];       s3 += cv * Wpc[HH + j];
        s4 += hv * Wu[j];
        s5 += hv * Wuse[j];      s6 += hv * Wuse[HH + j];
    }

    __shared__ float red[7][256];
    red[0][tid] = s0; red[1][tid] = s1; red[2][tid] = s2; red[3][tid] = s3;
    red[4][tid] = s4; red[5][tid] = s5; red[6][tid] = s6;
    __syncthreads();
    for (int off = 128; off > 0; off >>= 1) {
        if (tid < off) {
#pragma unroll
            for (int k = 0; k < 7; ++k) red[k][tid] += red[k][tid + off];
        }
        __syncthreads();
    }

    if (tid == 0) {
        float* out_pred  = out;
        float* out_predc = out + TT * BB * 2;
        float* out_util  = out + TT * BB * 4;
        float* out_watch = out + TT * BB * 5;

        int tb = t * BB + b;
        out_pred[tb * 2 + 0]  = red[0][0] + bp[0];
        out_pred[tb * 2 + 1]  = red[1][0] + bp[1];
        out_predc[tb * 2 + 0] = red[2][0] + bpc[0];
        out_predc[tb * 2 + 1] = red[3][0] + bpc[1];
        out_util[tb]          = red[4][0] + bu[0];

        // gumbel noise (counter-based hash; stands in for jax threefry)
        unsigned base = (unsigned)(tb * 2) * 0x9E3779B9u + 0x85EBCA6Bu;
        float u0 = u01_hash(base + 0u), u1 = u01_hash(base + 1u);
        float g0 = -logf(-logf(u0));
        float g1 = -logf(-logf(u1));
        float l0 = red[5][0] + buse[0] + g0;
        float l1 = red[6][0] + buse[1] + g1;
        float m = fmaxf(l0, l1);
        float e0 = expf(l0 - m), e1 = expf(l1 - m);
        float inv = 1.0f / (e0 + e1);
        if (t >= 1) {
            int wb = (t - 1) * BB + b;
            out_watch[wb * 2 + 0] = e0 * inv;
            out_watch[wb * 2 + 1] = e1 * inv;
        }
    }
}

// ---------------------------------------------------------------------------
// Host launcher
// ---------------------------------------------------------------------------
extern "C" void kernel_launch(void* const* d_in, const int* in_sizes, int n_in,
                              void* d_out, int out_size, void* d_ws, size_t ws_size,
                              hipStream_t stream) {
    const float* feature = (const float*)d_in[0];   // (B,T,F)
    const float* W_ih    = (const float*)d_in[1];   // (4H,F)
    const float* W_hh    = (const float*)d_in[2];   // (4H,H)
    const float* b_ih    = (const float*)d_in[3];
    const float* b_hh    = (const float*)d_in[4];
    const float* W_pred  = (const float*)d_in[5];   // (2,H)
    const float* b_pred  = (const float*)d_in[6];
    const float* W_predc = (const float*)d_in[7];   // (2,H)
    const float* b_predc = (const float*)d_in[8];
    const float* W_util  = (const float*)d_in[9];   // (1,H)
    const float* b_util  = (const float*)d_in[10];
    const float* W_use   = (const float*)d_in[11];  // (2,H)
    const float* b_use   = (const float*)d_in[12];
    float* out = (float*)d_out;

    // Workspace carve-up (all offsets >= 32B aligned); total ~97.5 MiB
    char* w = (char*)d_ws;
    __bf16* Wcat = (__bf16*)w;                 w += (size_t)GG * KK * 2;            // 64 MiB
    __bf16* xbf  = (__bf16*)w;                 w += (size_t)TT * BB * FF * 2;       // 24 MiB
    __bf16* hbf  = (__bf16*)w;                 w += (size_t)BB * HH * 2;            // 256 KiB
    float*  hbuf = (float*)w;                  w += (size_t)BB * HH * 4;            // 512 KiB
    float*  cbuf = (float*)w;                  w += (size_t)BB * HH * 4;            // 512 KiB
    float*  part = (float*)w;                  w += (size_t)KSPLIT * BB * GG * 4;   // 8 MiB
    float*  bsum = (float*)w;                  /* 32 KiB */

    // Phase 0: conversions + state init
    cvt_wcat_kernel<<<(GG * (size_t)KK) / 256, 256, 0, stream>>>(W_ih, W_hh, Wcat);
    cvt_x_kernel<<<((size_t)TT * BB * FF) / 256, 256, 0, stream>>>(feature, xbf);
    init_state_kernel<<<(BB * HH) / 256, 256, 0, stream>>>(b_ih, b_hh, bsum, hbuf, cbuf, hbf);

    // Phase 1: serial recurrence (96 steps, 2048 WMMA waves per step)
    for (int t = 0; t < TT; ++t) {
        const __bf16* xbf_t = xbf + (size_t)t * BB * FF;
        gemm_step_kernel<<<256, 256, 0, stream>>>(xbf_t, hbf, Wcat, part);
        cell_kernel<<<(BB * HH) / 256, 256, 0, stream>>>(part, bsum, hbuf, cbuf, hbf);
        heads_kernel<<<BB, 256, 0, stream>>>(hbuf, cbuf,
                                             W_pred, b_pred, W_predc, b_predc,
                                             W_util, b_util, W_use, b_use,
                                             out, t);
    }
}
